// MoELayer_71279277245024
// MI455X (gfx1250) — matrix operand order repaired and checked by reading, not compiled, not run on hardware
//
#include <hip/hip_runtime.h>
#include <hip/hip_bf16.h>

// ---------------------------------------------------------------------------
// MoE layer for MI455X (gfx1250, wave32).
// T=8192 tokens, H=1024, I=4096, E=8, top-2, cap=2560.
// ~515 GFLOP vs ~25us of HBM traffic -> compute bound -> bf16 WMMA path.
// Weights are pre-converted+transposed to bf16 once per expert so GEMM tile
// fills are pure b128 copies, done via the CDNA5 async Global->LDS engine
// (ASYNCcnt) with double-buffered LDS when the builtins are available.
// ---------------------------------------------------------------------------

#define TK    8192          // tokens (B*S)
#define HDIM  1024
#define IDIM  4096
#define NE    8
#define CAP   2560          // int(1.25 * ceil(T*K/E))

#if defined(__gfx1250__) && \
    __has_builtin(__builtin_amdgcn_global_load_async_to_lds_b128) && \
    __has_builtin(__builtin_amdgcn_s_wait_asynccnt)
#define USE_ASYNC 1
#else
#define USE_ASYNC 0
#endif

#if USE_ASYNC
#define WAITA(n) __builtin_amdgcn_s_wait_asynccnt(n)
#else
#define WAITA(n)
#endif

typedef __attribute__((ext_vector_type(16))) __bf16 v16bf;
typedef __attribute__((ext_vector_type(8)))  __bf16 v8bf;
typedef __attribute__((ext_vector_type(8)))  float  v8f;

// Pointer types matching the async-builtin signature:
// arg0 = v4i addrspace(1)* (global), arg1 = v4i addrspace(3)* (LDS).
typedef int gv4i __attribute__((vector_size(16)));
typedef __attribute__((address_space(1))) gv4i* gas1_p;
typedef __attribute__((address_space(3))) gv4i* gas3_p;

__device__ __forceinline__ v16bf cat16(v8bf lo, v8bf hi) {
  return __builtin_shufflevector(lo, hi, 0,1,2,3,4,5,6,7,8,9,10,11,12,13,14,15);
}

// 16B tile-fill: async Global->LDS DMA when available, else VGPR copy.
__device__ __forceinline__ void cp16(void* lds, const void* g) {
#if USE_ASYNC
  __builtin_amdgcn_global_load_async_to_lds_b128(
      (gas1_p)(unsigned long long)(uintptr_t)g,   // AS1: same numeric address
      (gas3_p)lds,                                // generic->local addrspacecast
      0, 0);
#else
  *(uint4*)lds = *(const uint4*)g;
#endif
}

// ---------------------------------------------------------------------------
// Router: one wave32 per token. logits = x[t,:] @ rg[H,E]; softmax; top-2.
// ---------------------------------------------------------------------------
__global__ __launch_bounds__(256)
void moe_router(const float* __restrict__ x,      // [T][H]
                const float* __restrict__ rg,     // [H][E]
                int*   __restrict__ topi,         // [T][2]
                float* __restrict__ topw)         // [T][2]
{
  const int lane = threadIdx.x & 31;
  const int wave = threadIdx.x >> 5;
  const int t = blockIdx.x * 8 + wave;

  float acc[NE] = {0.f,0.f,0.f,0.f,0.f,0.f,0.f,0.f};
  for (int k = lane; k < HDIM; k += 32) {
    const float xv = x[t * HDIM + k];
    const float4 r0 = *(const float4*)&rg[k * NE];
    const float4 r1 = *(const float4*)&rg[k * NE + 4];
    acc[0] += xv * r0.x; acc[1] += xv * r0.y; acc[2] += xv * r0.z; acc[3] += xv * r0.w;
    acc[4] += xv * r1.x; acc[5] += xv * r1.y; acc[6] += xv * r1.z; acc[7] += xv * r1.w;
  }
#pragma unroll
  for (int e = 0; e < NE; ++e) {
#pragma unroll
    for (int off = 16; off > 0; off >>= 1)
      acc[e] += __shfl_xor(acc[e], off, 32);
  }
  if (lane == 0) {
    float mx = acc[0];
#pragma unroll
    for (int e = 1; e < NE; ++e) mx = fmaxf(mx, acc[e]);
    float p[NE]; float sum = 0.f;
#pragma unroll
    for (int e = 0; e < NE; ++e) { p[e] = __expf(acc[e] - mx); sum += p[e]; }
    const float inv = 1.0f / sum;
#pragma unroll
    for (int e = 0; e < NE; ++e) p[e] *= inv;
    int i0 = 0;
#pragma unroll
    for (int e = 1; e < NE; ++e) if (p[e] > p[i0]) i0 = e;
    int i1 = (i0 == 0) ? 1 : 0;
#pragma unroll
    for (int e = 0; e < NE; ++e) if (e != i0 && p[e] > p[i1]) i1 = e;
    const float w0 = p[i0], w1 = p[i1];
    const float s = 1.0f / (w0 + w1 + 1e-9f);
    topi[t * 2 + 0] = i0;      topi[t * 2 + 1] = i1;
    topw[t * 2 + 0] = w0 * s;  topw[t * 2 + 1] = w1 * s;
  }
}

// ---------------------------------------------------------------------------
// Per-expert compaction: one wave per expert, deterministic token order via
// ballot + prefix popcount. Zero-fills padding slots up to CAP.
// ---------------------------------------------------------------------------
__global__ __launch_bounds__(32)
void moe_compact(const int*   __restrict__ topi,
                 const float* __restrict__ topw,
                 int*   __restrict__ tok,         // [E][CAP]
                 float* __restrict__ gw,          // [E][CAP]
                 int*   __restrict__ cnt)         // [E]
{
  const int e = blockIdx.x;
  const int lane = threadIdx.x;
  int base = 0;
  for (int t0 = 0; t0 < TK; t0 += 32) {
    const int t = t0 + lane;
    const int i0 = topi[t * 2 + 0];
    const int i1 = topi[t * 2 + 1];
    const bool sel0 = (i0 == e);
    const bool sel1 = (i1 == e);
    const bool pred = sel0 || sel1;
    const float w = sel0 ? topw[t * 2 + 0] : (sel1 ? topw[t * 2 + 1] : 0.f);
    const unsigned mask = (unsigned)__ballot(pred);   // wave32: low 32 bits
    const int prefix = __popc(mask & ((1u << lane) - 1u));
    const int slot = base + prefix;
    if (pred && slot < CAP) { tok[e * CAP + slot] = t; gw[e * CAP + slot] = w; }
    base += __popc(mask);
  }
  const int count = base < CAP ? base : CAP;
  for (int s = count + lane; s < CAP; s += 32) {
    tok[e * CAP + s] = 0; gw[e * CAP + s] = 0.f;
  }
  if (lane == 0) cnt[e] = count;
}

// ---------------------------------------------------------------------------
// Gather expert rows fp32 -> bf16; zero rows past count.
// ---------------------------------------------------------------------------
__global__ __launch_bounds__(256)
void moe_gather(const float* __restrict__ x,      // [T][H]
                const int*   __restrict__ tok,    // [CAP] (expert slice)
                const int*   __restrict__ cnt,    // scalar
                __bf16*      __restrict__ xe)     // [CAP][H]
{
  const int idx = (blockIdx.x * 256 + threadIdx.x) * 8;  // element in [CAP][H]
  const int s = idx >> 10;
  const int c = idx & 1023;
  v8bf v;
#pragma unroll
  for (int i = 0; i < 8; ++i) v[i] = (__bf16)0.f;
  if (s < *cnt) {
    const int t = tok[s];
    const float4 a = *(const float4*)&x[t * HDIM + c];
    const float4 b = *(const float4*)&x[t * HDIM + c + 4];
    v[0] = (__bf16)a.x; v[1] = (__bf16)a.y; v[2] = (__bf16)a.z; v[3] = (__bf16)a.w;
    v[4] = (__bf16)b.x; v[5] = (__bf16)b.y; v[6] = (__bf16)b.z; v[7] = (__bf16)b.w;
  }
  *(v8bf*)&xe[idx] = v;
}

// ---------------------------------------------------------------------------
// Transpose + convert: src fp32 [R][C] -> dst bf16 [C][R].
// Done once per expert weight matrix so GEMM B-tiles are contiguous [n][k]
// bf16 rows (exactly the CDNA5 B-fragment layout: lane = N, K as two 8-wide runs per half).
// ---------------------------------------------------------------------------
__global__ __launch_bounds__(256)
void moe_tcvt(const float* __restrict__ src, __bf16* __restrict__ dst,
              int R, int C)
{
  __shared__ float tile[32][33];
  const int tx = threadIdx.x & 31;
  const int ty = threadIdx.x >> 5;   // 0..7
  const int r0 = blockIdx.y * 32;
  const int c0 = blockIdx.x * 32;
#pragma unroll
  for (int i = 0; i < 4; ++i) {
    const int r = ty + i * 8;
    tile[r][tx] = src[(r0 + r) * C + c0 + tx];
  }
  __syncthreads();
#pragma unroll
  for (int i = 0; i < 4; ++i) {
    const int c = ty + i * 8;
    dst[(c0 + c) * R + r0 + tx] = (__bf16)tile[tx][c];
  }
}

// ---------------------------------------------------------------------------
// GEMM1 (fused gate+up+SiLU): h = silu(xe @ Wg) * (xe @ Wu), bf16 out.
// Block 128(M) x 128(N), BK=64; 8 waves as 2(M) x 4(N), wave tile 64x32 ->
// 16 wmma per 16 ds_load_b128 per K-step. A and both B tiles are plain
// 128x64 bf16 b128 copies (B pre-transposed), double-buffered + async.
// ---------------------------------------------------------------------------
__global__ __launch_bounds__(256)
void moe_gemm1(const __bf16* __restrict__ xe,     // [CAP][H]
               const __bf16* __restrict__ wgT,    // [I][H] = Wg^T bf16
               const __bf16* __restrict__ wuT,    // [I][H] = Wu^T bf16
               __bf16*       __restrict__ hbuf,   // [CAP][I]
               const int*    __restrict__ cnt)
{
  const int mBase = blockIdx.y * 128;
  if (mBase >= *cnt) return;                 // uniform early-exit past count
  const int nBase = blockIdx.x * 128;

  __shared__ __align__(16) __bf16 As[2][128][72];   // [m][k]
  __shared__ __align__(16) __bf16 Bg[2][128][72];   // [n][k]
  __shared__ __align__(16) __bf16 Bu[2][128][72];   // [n][k]

  const int tid  = threadIdx.x;
  const int lane = tid & 31;
  const int wave = tid >> 5;
  const int mW   = (wave & 1) * 64;          // 2 waves in M
  const int nW   = (wave >> 1) * 32;         // 4 waves in N
  const int half = lane >> 4;
  const int l16  = lane & 15;

  v8f accG[4][2] = {};
  v8f accU[4][2] = {};

  // ---- stage fill: 4 rounds x 3 arrays = 12 async copies per thread ----
  auto issue = [&](int k0, int b) {
#pragma unroll
    for (int i = 0; i < 4; ++i) {
      const int u   = tid + i * 256;         // 0..1023
      const int row = u >> 3;                // 0..127
      const int col = (u & 7) * 8;           // 0,8,..,56
      cp16(&As[b][row][col], &xe [(mBase + row) * HDIM + k0 + col]);
      cp16(&Bg[b][row][col], &wgT[(nBase + row) * HDIM + k0 + col]);
      cp16(&Bu[b][row][col], &wuT[(nBase + row) * HDIM + k0 + col]);
    }
  };

  auto compute = [&](int b) {
#pragma unroll
    for (int kk = 0; kk < 64; kk += 32) {
      v16bf a[4];
#pragma unroll
      for (int i = 0; i < 4; ++i) {          // A frag: row/lane, K split by half
        const int row = mW + i * 16 + l16;
        const int o0  = kk + half * 8;
        a[i] = cat16(*(const v8bf*)&As[b][row][o0],
                     *(const v8bf*)&As[b][row][o0 + 16]);
      }
#pragma unroll
      for (int j = 0; j < 2; ++j) {          // B frag: N=lane&15, K split by half (same as A)
        const int nc = nW + j * 16 + l16;
        const int kr = kk + half * 8;
        const v16bf bg = cat16(*(const v8bf*)&Bg[b][nc][kr],
                               *(const v8bf*)&Bg[b][nc][kr + 16]);
        const v16bf bu = cat16(*(const v8bf*)&Bu[b][nc][kr],
                               *(const v8bf*)&Bu[b][nc][kr + 16]);
#pragma unroll
        for (int i = 0; i < 4; ++i) {
          accG[i][j] = __builtin_amdgcn_wmma_f32_16x16x32_bf16(
              false, a[i], false, bg, (short)0, accG[i][j], false, false);
          accU[i][j] = __builtin_amdgcn_wmma_f32_16x16x32_bf16(
              false, a[i], false, bu, (short)0, accU[i][j], false, false);
        }
      }
    }
  };

  const int NSTAGE = HDIM / 64;              // 16
  int buf = 0;
  issue(0, 0);
  for (int s = 0; s < NSTAGE - 1; ++s) {
    issue((s + 1) * 64, buf ^ 1);
    WAITA(12);                               // stage s landed (in-order)
    __syncthreads();
    compute(buf);
    __syncthreads();
    buf ^= 1;
  }
  WAITA(0);
  __syncthreads();
  compute(buf);

  // Epilogue: SwiGLU, convert, coalesced-across-lanes b16 stores.
#pragma unroll
  for (int i = 0; i < 4; ++i) {
#pragma unroll
    for (int j = 0; j < 2; ++j) {
      const int col = nBase + nW + j * 16 + l16;
#pragma unroll
      for (int v = 0; v < 8; ++v) {
        const int row = mBase + mW + i * 16 + half * 8 + v;
        const float g = accG[i][j][v];
        const float u = accU[i][j][v];
        const float sg = g / (1.0f + __expf(-g));
        hbuf[row * IDIM + col] = (__bf16)(sg * u);
      }
    }
  }
}

// ---------------------------------------------------------------------------
// GEMM2 (down proj): y[tok[s]] += gate[s] * (h[s] @ Wd). Block 128x128,
// K=4096 in 64 double-buffered async stages. Experts serialize on the
// stream and each (token,col) is touched once per expert -> deterministic.
// ---------------------------------------------------------------------------
__global__ __launch_bounds__(256)
void moe_gemm2(const __bf16* __restrict__ hbuf,   // [CAP][I]
               const __bf16* __restrict__ wdT,    // [H][I] = Wd^T bf16
               const int*    __restrict__ tok,    // [CAP] expert slice
               const float*  __restrict__ gw,     // [CAP] expert slice
               const int*    __restrict__ cnt,
               float*        __restrict__ out)    // [T][H]
{
  const int mBase = blockIdx.y * 128;
  if (mBase >= *cnt) return;
  const int nBase = blockIdx.x * 128;

  __shared__ __align__(16) __bf16 As[2][128][72];
  __shared__ __align__(16) __bf16 Bd[2][128][72];
  __shared__ int   s_tok[128];
  __shared__ float s_gate[128];

  const int tid  = threadIdx.x;
  const int lane = tid & 31;
  const int wave = tid >> 5;
  const int mW   = (wave & 1) * 64;
  const int nW   = (wave >> 1) * 32;
  const int half = lane >> 4;
  const int l16  = lane & 15;

  if (tid < 128) { s_tok[tid] = tok[mBase + tid]; s_gate[tid] = gw[mBase + tid]; }

  v8f acc[4][2] = {};

  auto issue = [&](int k0, int b) {
#pragma unroll
    for (int i = 0; i < 4; ++i) {
      const int u   = tid + i * 256;
      const int row = u >> 3;
      const int col = (u & 7) * 8;
      cp16(&As[b][row][col], &hbuf[(mBase + row) * IDIM + k0 + col]);
      cp16(&Bd[b][row][col], &wdT [(nBase + row) * IDIM + k0 + col]);
    }
  };

  auto compute = [&](int b) {
#pragma unroll
    for (int kk = 0; kk < 64; kk += 32) {
      v16bf a[4];
#pragma unroll
      for (int i = 0; i < 4; ++i) {
        const int row = mW + i * 16 + l16;
        const int o0  = kk + half * 8;
        a[i] = cat16(*(const v8bf*)&As[b][row][o0],
                     *(const v8bf*)&As[b][row][o0 + 16]);
      }
#pragma unroll
      for (int j = 0; j < 2; ++j) {
        const int nc = nW + j * 16 + l16;
        const int kr = kk + half * 8;
        const v16bf bd = cat16(*(const v8bf*)&Bd[b][nc][kr],
                               *(const v8bf*)&Bd[b][nc][kr + 16]);
#pragma unroll
        for (int i = 0; i < 4; ++i) {
          acc[i][j] = __builtin_amdgcn_wmma_f32_16x16x32_bf16(
              false, a[i], false, bd, (short)0, acc[i][j], false, false);
        }
      }
    }
  };

  const int NSTAGE = IDIM / 64;              // 64
  int buf = 0;
  issue(0, 0);
  for (int s = 0; s < NSTAGE - 1; ++s) {
    issue((s + 1) * 64, buf ^ 1);
    WAITA(8);
    __syncthreads();
    compute(buf);
    __syncthreads();
    buf ^= 1;
  }
  WAITA(0);
  __syncthreads();
  compute(buf);

  // Epilogue: scale by gate, scatter-add into out.
#pragma unroll
  for (int i = 0; i < 4; ++i) {
#pragma unroll
    for (int j = 0; j < 2; ++j) {
      const int col = nBase + nW + j * 16 + l16;
#pragma unroll
      for (int v = 0; v < 8; ++v) {
        const int rloc = mW + i * 16 + half * 8 + v;
        const float gate = s_gate[rloc];
        if (gate != 0.f) {
          const int t = s_tok[rloc];
          atomicAdd(&out[t * HDIM + col], acc[i][j][v] * gate);
        }
      }
    }
  }
}

// ---------------------------------------------------------------------------
// Host-side launcher. Experts run sequentially on `stream` so the xe (5MB),
// h (20MB) and transposed bf16 weight (3 x 8MB) buffers are reused.
// Total workspace ~52 MB.
// ---------------------------------------------------------------------------
extern "C" void kernel_launch(void* const* d_in, const int* in_sizes, int n_in,
                              void* d_out, int out_size, void* d_ws, size_t ws_size,
                              hipStream_t stream) {
  const float* x      = (const float*)d_in[0];   // [4,2048,1024]
  const float* rg     = (const float*)d_in[1];   // [1024,8]
  const float* gate_w = (const float*)d_in[2];   // [8,1024,4096]
  const float* up_w   = (const float*)d_in[3];   // [8,1024,4096]
  const float* down_w = (const float*)d_in[4];   // [8,4096,1024]
  float* out = (float*)d_out;
  (void)out;

  char* ws = (char*)d_ws;
  size_t off = 0;
  int*    topi = (int*)(ws + off);   off += (size_t)TK * 2 * 4;          // 64 KB
  float*  topw = (float*)(ws + off); off += (size_t)TK * 2 * 4;          // 64 KB
  int*    tok  = (int*)(ws + off);   off += (size_t)NE * CAP * 4;        // 80 KB
  float*  gw   = (float*)(ws + off); off += (size_t)NE * CAP * 4;        // 80 KB
  int*    cnt  = (int*)(ws + off);   off += 256;                         // 32 B
  __bf16* xe   = (__bf16*)(ws + off); off += (size_t)CAP * HDIM * 2;     //  5 MB
  __bf16* hb   = (__bf16*)(ws + off); off += (size_t)CAP * IDIM * 2;     // 20 MB
  __bf16* wgT  = (__bf16*)(ws + off); off += (size_t)IDIM * HDIM * 2;    //  8 MB
  __bf16* wuT  = (__bf16*)(ws + off); off += (size_t)IDIM * HDIM * 2;    //  8 MB
  __bf16* wdT  = (__bf16*)(ws + off); off += (size_t)IDIM * HDIM * 2;    //  8 MB

  (void)hipMemsetAsync(d_out, 0, (size_t)TK * HDIM * sizeof(float), stream);

  moe_router <<<TK / 8, 256, 0, stream>>>(x, rg, topi, topw);
  moe_compact<<<NE, 32, 0, stream>>>(topi, topw, tok, gw, cnt);

  for (int e = 0; e < NE; ++e) {
    moe_gather<<<(CAP * HDIM / 8) / 256, 256, 0, stream>>>(
        x, tok + e * CAP, cnt + e, xe);

    // Wg,Wu: [H][I] -> [I][H];  Wd: [I][H] -> [H][I]   (fp32 -> bf16)
    moe_tcvt<<<dim3(IDIM / 32, HDIM / 32), 256, 0, stream>>>(
        gate_w + (size_t)e * HDIM * IDIM, wgT, HDIM, IDIM);
    moe_tcvt<<<dim3(IDIM / 32, HDIM / 32), 256, 0, stream>>>(
        up_w   + (size_t)e * HDIM * IDIM, wuT, HDIM, IDIM);
    moe_tcvt<<<dim3(HDIM / 32, IDIM / 32), 256, 0, stream>>>(
        down_w + (size_t)e * IDIM * HDIM, wdT, IDIM, HDIM);

    dim3 g1(IDIM / 128, CAP / 128);
    moe_gemm1<<<g1, 256, 0, stream>>>(xe, wgT, wuT, hb, cnt + e);

    dim3 g2(HDIM / 128, CAP / 128);
    moe_gemm2<<<g2, 256, 0, stream>>>(hb, wdT, tok + e * CAP, gw + e * CAP,
                                      cnt + e, out);
  }
}
